// ConformerBlock_11544872092209
// MI455X (gfx1250) — matrix operand order repaired and checked by reading, not compiled, not run on hardware
//
#include <hip/hip_runtime.h>
#include <cstdint>
#include <cstddef>

typedef __attribute__((ext_vector_type(16))) __bf16 v16bf;
typedef __attribute__((ext_vector_type(8)))  __bf16 v8bf;
typedef __attribute__((ext_vector_type(8)))  float  v8f;
typedef __attribute__((ext_vector_type(4)))  unsigned int u32x4;
typedef __attribute__((ext_vector_type(4)))  int i32x4;
typedef __attribute__((ext_vector_type(8)))  int i32x8;

#define DEV static __device__ __forceinline__

static constexpr int Bc = 8, Tc = 1024, Dc = 512, NHc = 8, HDc = 64, Hh = 127, Kc = 32;
static constexpr int Mrows = Bc * Tc;           // 8192

// Tensor Data Mover available? (device pass only; arity differs per toolchain)
#if defined(__HIP_DEVICE_COMPILE__) && defined(__gfx1250__) && __has_builtin(__builtin_amdgcn_tensor_load_to_lds)
#define USE_TDM 1
#else
#define USE_TDM 0
#endif

DEV v8f wmma_bf16(v16bf a, v16bf b, v8f c) {
  return __builtin_amdgcn_wmma_f32_16x16x32_bf16(false, a, false, b, (short)0, c, false, false);
}

// A fragment 16x32 bf16 from LDS tile [row][k] stride 32.
// lane&15 = row ; K elements {c0..c0+15}, c0 = (lane>>4)*16
DEV v16bf load_fragA_lds(const __bf16* base, int lane) {
  const int r = lane & 15, c0 = (lane >> 4) * 16;
  const __bf16* p = base + r * 32 + c0;
  v8bf lo = *(const v8bf*)p;
  v8bf hi = *(const v8bf*)(p + 8);
  v16bf f;
#pragma unroll
  for (int i = 0; i < 8; ++i) { f[i] = lo[i]; f[i + 8] = hi[i]; }
  return f;
}

// B fragment 32x16 bf16 from LDS tile stored transposed [n][k] stride 32.
// lane&15 = n ; contiguous 16 K values at k0 = (lane>>4)*16
DEV v16bf load_fragB_lds(const __bf16* base, int lane) {
  const int n = lane & 15, k0 = (lane >> 4) * 16;
  return *(const v16bf*)(base + n * 32 + k0);
}

#if USE_TDM
// Issue a 2D TDM tile load: tile (tl0 x tl1) bf16 elements from a row-major
// tensor (row length td0 elems, td1 rows, row stride s0 elems) into LDS.
// D# layout per CDNA5 ISA ch.8 (group0: count/lds/global/type, group1: dims).
DEV void tdm_load_tile_2d(const void* gaddr, const void* lds_ptr,
                          int td0, int td1, int tl0, int tl1, long long s0) {
  const unsigned long long ga = (unsigned long long)(uintptr_t)gaddr;
  const unsigned lds_off = (unsigned)(unsigned long long)(uintptr_t)lds_ptr;
  u32x4 g0;
  g0[0] = 1u;                                            // count=1, user desc
  g0[1] = lds_off;                                       // lds_addr (bytes)
  g0[2] = (unsigned)(ga & 0xFFFFFFFFu);                  // global_addr[31:0]
  g0[3] = ((unsigned)((ga >> 32) & 0x01FFFFFFu)) | 0x80000000u; // [56:32] | type=2
  i32x8 g1;
  g1[0] = (1 << 16);                                     // data_size=1 (2 bytes)
  g1[1] = (td0 & 0xFFFF) << 16;                          // tensor_dim0[15:0]
  g1[2] = ((td0 >> 16) & 0xFFFF) | ((td1 & 0xFFFF) << 16);
  g1[3] = ((td1 >> 16) & 0xFFFF) | ((tl0 & 0xFFFF) << 16); // tile_dim0
  g1[4] = (tl1 & 0xFFFF);                                // tile_dim1 (tile_dim2=0)
  g1[5] = (int)(s0 & 0xFFFFFFFFll);                      // dim0_stride[31:0]
  g1[6] = (int)((s0 >> 32) & 0xFFFFll);                  // dim0_stride[47:32]
  g1[7] = 0;
  i32x4 z4 = {0, 0, 0, 0};
#if __clang_major__ >= 23
  i32x8 z8 = {0, 0, 0, 0, 0, 0, 0, 0};
  __builtin_amdgcn_tensor_load_to_lds(g0, g1, z4, z4, z8, 0);
#else
  __builtin_amdgcn_tensor_load_to_lds(g0, g1, z4, z4, 0);
#endif
}
#endif

// ---------------------------------------------------------------------------
// Tiled GEMM: out[M,N] = epilogue(A[M,K](bf16) @ W[K,N](bf16) + bias)
// 256 threads (8 waves), 128x128 block tile, BLK_K=32, double-buffered LDS.
// Wave tile 32x64 -> 8 WMMAs per wave per k-step.
// A tile staged by the Tensor Data Mover (TENSORcnt), W tile staged manually
// (transposed) through registers. EPI:
//   0 = f32 out (+bias)
//   1 = SiLU(+bias) -> bf16 out
//   2 = bf16 out (+bias if non-null)
//   3 = f32 out = res + 0.5*(+bias), masked rows zeroed
//   4 = f32 out = res + (+bias), masked rows zeroed
// ---------------------------------------------------------------------------
template<int EPI>
__global__ __launch_bounds__(256) void gemm_bf16_kernel(
    const __bf16* __restrict__ A, const __bf16* __restrict__ W,
    const float* __restrict__ bias, const float* __restrict__ res,
    const unsigned char* __restrict__ mask,
    float* __restrict__ outf, __bf16* __restrict__ outb,
    int M, int N, int K)
{
  __shared__ __align__(32) __bf16 sA[2][128 * 32];
  __shared__ __align__(32) __bf16 sB[2][128 * 32];
  const int tid  = threadIdx.x;
  const int lane = tid & 31, wave = tid >> 5;
  const int wm = wave >> 1, wn = wave & 1;           // 4 x 2 wave grid
  const int m0 = blockIdx.y * 128, n0 = blockIdx.x * 128;

  v8f acc[2][4];
#pragma unroll
  for (int i = 0; i < 2; ++i)
#pragma unroll
    for (int j = 0; j < 4; ++j)
#pragma unroll
      for (int r = 0; r < 8; ++r) acc[i][j][r] = 0.0f;

  const int arow = tid >> 1, acol = (tid & 1) * 16;  // A fallback loader map
  const int wkr = tid >> 4, wn8 = (tid & 15) * 8;    // W loader map

  v8bf wreg[2];
#if !USE_TDM
  v8bf areg[2];
#endif

  // ---- prologue: stage k0 = 0 into buffer 0 ----
#if USE_TDM
  if (wave == 0)
    tdm_load_tile_2d(A + (size_t)m0 * K, &sA[0][0], K, M, 32, 128, (long long)K);
#else
  {
    const __bf16* src = A + (size_t)(m0 + arow) * K + acol;
    *(v8bf*)(&sA[0][arow * 32 + acol])     = *(const v8bf*)src;
    *(v8bf*)(&sA[0][arow * 32 + acol + 8]) = *(const v8bf*)(src + 8);
  }
#endif
#pragma unroll
  for (int p = 0; p < 2; ++p)
    wreg[p] = *(const v8bf*)(W + (size_t)(wkr + p * 16) * N + n0 + wn8);
#pragma unroll
  for (int p = 0; p < 2; ++p)
#pragma unroll
    for (int i = 0; i < 8; ++i) sB[0][(wn8 + i) * 32 + wkr + p * 16] = wreg[p][i];
#if USE_TDM
  if (wave == 0) __builtin_amdgcn_s_wait_tensorcnt(0);
#endif
  __syncthreads();

  int buf = 0;
  for (int k0 = 0; k0 < K; k0 += 32, buf ^= 1) {
    const int nb = buf ^ 1;
    const bool more = (k0 + 32 < K);
    if (more) {
      // kick off next A tile DMA / next W tile global loads
#if USE_TDM
      if (wave == 0)
        tdm_load_tile_2d(A + (size_t)m0 * K + (k0 + 32), &sA[nb][0],
                         K, M, 32, 128, (long long)K);
#else
      const __bf16* src = A + (size_t)(m0 + arow) * K + k0 + 32 + acol;
      areg[0] = *(const v8bf*)src;
      areg[1] = *(const v8bf*)(src + 8);
#endif
#pragma unroll
      for (int p = 0; p < 2; ++p)
        wreg[p] = *(const v8bf*)(W + (size_t)(k0 + 32 + wkr + p * 16) * N + n0 + wn8);
      if (k0 + 64 < K)
        __builtin_prefetch(W + (size_t)(k0 + 64 + wkr) * N + n0 + wn8, 0, 1);
    }

    // ---- compute on current buffer ----
    v16bf aF[2], bF[4];
#pragma unroll
    for (int i = 0; i < 2; ++i)
      aF[i] = load_fragA_lds(&sA[buf][(wm * 32 + i * 16) * 32], lane);
#pragma unroll
    for (int j = 0; j < 4; ++j)
      bF[j] = load_fragB_lds(&sB[buf][(wn * 64 + j * 16) * 32], lane);
#pragma unroll
    for (int i = 0; i < 2; ++i)
#pragma unroll
      for (int j = 0; j < 4; ++j) acc[i][j] = wmma_bf16(aF[i], bF[j], acc[i][j]);

    if (more) {
#pragma unroll
      for (int p = 0; p < 2; ++p)
#pragma unroll
        for (int i = 0; i < 8; ++i) sB[nb][(wn8 + i) * 32 + wkr + p * 16] = wreg[p][i];
#if USE_TDM
      if (wave == 0) __builtin_amdgcn_s_wait_tensorcnt(0);
#else
      *(v8bf*)(&sA[nb][arow * 32 + acol])     = areg[0];
      *(v8bf*)(&sA[nb][arow * 32 + acol + 8]) = areg[1];
#endif
    }
    __syncthreads();
  }

  // ---- epilogue ----
#pragma unroll
  for (int i = 0; i < 2; ++i)
#pragma unroll
    for (int j = 0; j < 4; ++j)
#pragma unroll
      for (int r = 0; r < 8; ++r) {
        const int m = m0 + wm * 32 + i * 16 + (lane >> 4) * 8 + r;
        const int n = n0 + wn * 64 + j * 16 + (lane & 15);
        const size_t idx = (size_t)m * N + n;
        float v = acc[i][j][r];
        if (bias) v += bias[n];
        if (EPI == 0) {
          outf[idx] = v;
        } else if (EPI == 1) {
          const float sg = 1.0f / (1.0f + __expf(-v));
          outb[idx] = (__bf16)(v * sg);
        } else if (EPI == 2) {
          outb[idx] = (__bf16)v;
        } else if (EPI == 3) {
          float o = res[idx] + 0.5f * v;
          if (mask[m]) o = 0.0f;
          outf[idx] = o;
        } else {
          float o = res[idx] + v;
          if (mask[m]) o = 0.0f;
          outf[idx] = o;
        }
      }
}

// ---------------------------------------------------------------------------
// LayerNorm over D=512. OUT=0: bf16 out. OUT=1: f32 out with mask-fill.
// ---------------------------------------------------------------------------
template<int OUT>
__global__ __launch_bounds__(256) void ln_kernel(
    const float* __restrict__ in, const float* __restrict__ g, const float* __restrict__ b,
    __bf16* __restrict__ outb, float* __restrict__ outf,
    const unsigned char* __restrict__ mask)
{
  __shared__ float red[256];
  const int row = blockIdx.x, tid = threadIdx.x;
  const float* r = in + (size_t)row * Dc;
  const float v0 = r[tid], v1 = r[tid + 256];
  red[tid] = v0 + v1;
  __syncthreads();
  for (int s = 128; s > 0; s >>= 1) { if (tid < s) red[tid] += red[tid + s]; __syncthreads(); }
  const float mean = red[0] * (1.0f / (float)Dc);
  __syncthreads();
  const float d0 = v0 - mean, d1 = v1 - mean;
  red[tid] = d0 * d0 + d1 * d1;
  __syncthreads();
  for (int s = 128; s > 0; s >>= 1) { if (tid < s) red[tid] += red[tid + s]; __syncthreads(); }
  const float rstd = rsqrtf(red[0] * (1.0f / (float)Dc) + 1e-5f);
  const float o0 = d0 * rstd * g[tid] + b[tid];
  const float o1 = d1 * rstd * g[tid + 256] + b[tid + 256];
  if (OUT == 0) {
    outb[(size_t)row * Dc + tid]       = (__bf16)o0;
    outb[(size_t)row * Dc + tid + 256] = (__bf16)o1;
  } else {
    const bool mk = mask[row] != 0;
    outf[(size_t)row * Dc + tid]       = mk ? 0.0f : o0;
    outf[(size_t)row * Dc + tid + 256] = mk ? 0.0f : o1;
  }
}

// ---------------------------------------------------------------------------
// Attention: one wave per (16-query tile, head, batch). Horizon H=127 =>
// key window = [q0-128, q0+32) -> 160 score columns (9 valid 16-key tiles).
// QK^T, Q*RelPos^T and P*V all via v_wmma_f32_16x16x32_bf16.
// qkv layout: [8192, 1536] bf16 (q | k | v, head-major 64 each).
// relb layout: [NH][128][64] bf16. out: [8192, 512] bf16 (head-major cols).
// ---------------------------------------------------------------------------
__global__ __launch_bounds__(32) void attn_kernel(
    const __bf16* __restrict__ qkv, const __bf16* __restrict__ relb,
    const unsigned char* __restrict__ mask, __bf16* __restrict__ outb)
{
  const int qt = blockIdx.x;          // 0..63
  const int h  = blockIdx.y;
  const int b  = blockIdx.z;
  const int lane = threadIdx.x;
  const int q0 = qt * 16;
  const float scale = 0.125f;         // 1/sqrt(64)

  __shared__ float qr[16 * 128];      // Q @ RelPos^T, per q-row bias table
  __shared__ float sS[16 * 160];      // score / prob buffer
  __shared__ __align__(32) __bf16 vt[64 * 32];  // V chunk transposed [d][key]

  // Q fragments (d chunks 0..31, 32..63), loaded straight from global.
  v16bf qa[2];
  {
    const __bf16* qrow = qkv + (size_t)(b * Tc + q0 + (lane & 15)) * (3 * Dc) + h * HDc;
    const int c0 = (lane >> 4) * 16;
#pragma unroll
    for (int ch = 0; ch < 2; ++ch) {
      const __bf16* p = qrow + ch * 32 + c0;
      v8bf lo = *(const v8bf*)p;
      v8bf hi = *(const v8bf*)(p + 8);
#pragma unroll
      for (int i = 0; i < 8; ++i) { qa[ch][i] = lo[i]; qa[ch][i + 8] = hi[i]; }
    }
  }

  // Qr[q][p] = Q[q] . rel_pos[p]  (8 p-tiles of 16)
#pragma unroll
  for (int pt = 0; pt < 8; ++pt) {
    v8f acc;
#pragma unroll
    for (int r = 0; r < 8; ++r) acc[r] = 0.0f;
#pragma unroll
    for (int ch = 0; ch < 2; ++ch) {
      v16bf bf = *(const v16bf*)(relb +
          ((size_t)(h * 128 + pt * 16 + (lane & 15))) * HDc + ch * 32 + (lane >> 4) * 16);
      acc = wmma_bf16(qa[ch], bf, acc);
    }
#pragma unroll
    for (int r = 0; r < 8; ++r)
      qr[((lane >> 4) * 8 + r) * 128 + pt * 16 + (lane & 15)] = acc[r];
  }

  // init all 160 score columns to -inf (cols 144..159 stay 0-prob pads)
  for (int i = 0; i < 80; ++i) sS[lane * 80 + i] = -__builtin_inff();
  __syncthreads();

  // scores for 9 key tiles kt = qt-8 .. qt
  for (int j = 0; j < 9; ++j) {
    const int kt = qt - 8 + j;
    if (kt < 0) continue;
    v8f acc;
#pragma unroll
    for (int r = 0; r < 8; ++r) acc[r] = 0.0f;
#pragma unroll
    for (int ch = 0; ch < 2; ++ch) {
      v16bf kf = *(const v16bf*)(qkv +
          (size_t)(b * Tc + kt * 16 + (lane & 15)) * (3 * Dc) + Dc + h * HDc +
          ch * 32 + (lane >> 4) * 16);
      acc = wmma_bf16(qa[ch], kf, acc);
    }
#pragma unroll
    for (int r = 0; r < 8; ++r) {
      const int qrow = (lane >> 4) * 8 + r;
      const int kcol = lane & 15;
      const int q = q0 + qrow, key = kt * 16 + kcol;
      const int p = q - key;
      float val = -__builtin_inff();
      if (p >= 0 && p <= Hh && !mask[b * Tc + key])
        val = (acc[r] + qr[qrow * 128 + p]) * scale;
      sS[qrow * 160 + j * 16 + kcol] = val;
    }
  }
  __syncthreads();

  // softmax over 160 columns: 2 lanes per row, 80 columns each
  {
    const int row = lane & 15, half = lane >> 4;
    float* sr = sS + row * 160 + half * 80;
    float mx = -__builtin_inff();
    for (int i = 0; i < 80; ++i) mx = fmaxf(mx, sr[i]);
    mx = fmaxf(mx, __shfl_xor(mx, 16));
    float sum = 0.0f;
    for (int i = 0; i < 80; ++i) { const float e = __expf(sr[i] - mx); sr[i] = e; sum += e; }
    sum += __shfl_xor(sum, 16);
    const float inv = 1.0f / sum;
    for (int i = 0; i < 80; ++i) sr[i] *= inv;
  }
  __syncthreads();

  // P @ V : 5 key chunks of 32, 4 d-tiles of 16
  v8f o[4];
#pragma unroll
  for (int nt = 0; nt < 4; ++nt)
#pragma unroll
    for (int r = 0; r < 8; ++r) o[nt][r] = 0.0f;

  const int win_lo = q0 - 128;
  for (int c = 0; c < 5; ++c) {
    { // stage V chunk transposed: vt[d][key_in_chunk]
      int key = win_lo + c * 32 + lane;
      key = key < 0 ? 0 : (key > Tc - 1 ? Tc - 1 : key);
      const __bf16* vr = qkv + (size_t)(b * Tc + key) * (3 * Dc) + 2 * Dc + h * HDc;
#pragma unroll
      for (int ch = 0; ch < 8; ++ch) {
        v8bf u = *(const v8bf*)(vr + ch * 8);
#pragma unroll
        for (int i = 0; i < 8; ++i) vt[(ch * 8 + i) * 32 + lane] = u[i];
      }
    }
    __syncthreads();

    // A fragment from probabilities (f32 -> bf16)
    v16bf af;
    {
      const int qrw = lane & 15, c0 = (lane >> 4) * 16;
      const float* sp = sS + qrw * 160 + c * 32 + c0;
#pragma unroll
      for (int e = 0; e < 8; ++e) { af[e] = (__bf16)sp[e]; af[e + 8] = (__bf16)sp[e + 8]; }
    }
#pragma unroll
    for (int nt = 0; nt < 4; ++nt) {
      v16bf vf = load_fragB_lds(vt + (nt * 16) * 32, lane);
      o[nt] = wmma_bf16(af, vf, o[nt]);
    }
    __syncthreads();
  }

  // store: out[b, q, h*64 + d] bf16
#pragma unroll
  for (int nt = 0; nt < 4; ++nt)
#pragma unroll
    for (int r = 0; r < 8; ++r)
      outb[(size_t)(b * Tc + q0 + (lane >> 4) * 8 + r) * Dc + h * HDc + nt * 16 + (lane & 15)] =
          (__bf16)o[nt][r];
}

// ---------------------------------------------------------------------------
// Small elementwise kernels
// ---------------------------------------------------------------------------
__global__ __launch_bounds__(256) void cvt_bf16_kernel(const float* __restrict__ in,
                                                       __bf16* __restrict__ out, int n) {
  const int i = blockIdx.x * 256 + threadIdx.x;
  if (i < n) out[i] = (__bf16)in[i];
}

// rel_pos [128, 8, 64] f32 -> [8][128][64] bf16
__global__ __launch_bounds__(256) void relpos_cvt_kernel(const float* __restrict__ in,
                                                         __bf16* __restrict__ out) {
  const int i = blockIdx.x * 256 + threadIdx.x;
  if (i >= 128 * 8 * 64) return;
  const int d = i & 63, ph = i >> 6;
  const int h = ph & 7, p = ph >> 3;
  out[((size_t)h * 128 + p) * 64 + d] = (__bf16)in[((size_t)p * 8 + h) * 64 + d];
}

__global__ __launch_bounds__(256) void glu_kernel(const float* __restrict__ in,
                                                  float* __restrict__ out) {
  const int i = blockIdx.x * 256 + threadIdx.x;
  if (i >= Mrows * Dc) return;
  const int c = i & (Dc - 1);
  const int row = i >> 9;
  const float a = in[(size_t)row * (2 * Dc) + c];
  const float g = in[(size_t)row * (2 * Dc) + Dc + c];
  out[i] = a * (1.0f / (1.0f + __expf(-g)));
}

__global__ __launch_bounds__(256) void dwconv_bn_silu_kernel(
    const float* __restrict__ glu, const float* __restrict__ dww, const float* __restrict__ dwb,
    const float* __restrict__ bng, const float* __restrict__ bnb,
    const float* __restrict__ bnm, const float* __restrict__ bnv,
    __bf16* __restrict__ out)
{
  const int i = blockIdx.x * 256 + threadIdx.x;
  if (i >= Mrows * Dc) return;
  const int d = i & (Dc - 1);
  const int row = i >> 9;
  const int t = row & (Tc - 1);
  const int base = row - t;   // b*T
  float acc = 0.0f;
#pragma unroll
  for (int j = 0; j < Kc; ++j) {
    const int tj = t - (Kc - 1) + j;
    const float xv = (tj >= 0) ? glu[(size_t)(base + tj) * Dc + d] : 0.0f;
    acc += xv * dww[d * Kc + j];
  }
  acc += dwb[d];
  acc = (acc - bnm[d]) * rsqrtf(bnv[d] + 1e-5f) * bng[d] + bnb[d];
  const float sg = 1.0f / (1.0f + __expf(-acc));
  out[i] = (__bf16)(acc * sg);
}

// ---------------------------------------------------------------------------
// Host orchestration
// ---------------------------------------------------------------------------
extern "C" void kernel_launch(void* const* d_in, const int* in_sizes, int n_in,
                              void* d_out, int out_size, void* d_ws, size_t ws_size,
                              hipStream_t stream) {
  (void)in_sizes; (void)n_in; (void)out_size; (void)ws_size;
  const float* x        = (const float*)d_in[0];
  const unsigned char* mask = (const unsigned char*)d_in[1];
  const float* ff1_ng   = (const float*)d_in[2];
  const float* ff1_nb   = (const float*)d_in[3];
  const float* ff1_w1   = (const float*)d_in[4];
  const float* ff1_b1   = (const float*)d_in[5];
  const float* ff1_w2   = (const float*)d_in[6];
  const float* ff1_b2   = (const float*)d_in[7];
  const float* mha_ng   = (const float*)d_in[8];
  const float* mha_nb   = (const float*)d_in[9];
  const float* wqkv     = (const float*)d_in[10];
  const float* wout     = (const float*)d_in[11];
  const float* bout     = (const float*)d_in[12];
  const float* rel_pos  = (const float*)d_in[13];
  const float* conv_ng  = (const float*)d_in[14];
  const float* conv_nb  = (const float*)d_in[15];
  const float* pw1_w    = (const float*)d_in[16];
  const float* pw1_b    = (const float*)d_in[17];
  const float* dw_w     = (const float*)d_in[18];
  const float* dw_b     = (const float*)d_in[19];
  const float* bn_g     = (const float*)d_in[20];
  const float* bn_b     = (const float*)d_in[21];
  const float* bn_m     = (const float*)d_in[22];
  const float* bn_v     = (const float*)d_in[23];
  const float* pw2_w    = (const float*)d_in[24];
  const float* pw2_b    = (const float*)d_in[25];
  const float* ff2_ng   = (const float*)d_in[26];
  const float* ff2_nb   = (const float*)d_in[27];
  const float* ff2_w1   = (const float*)d_in[28];
  const float* ff2_b1   = (const float*)d_in[29];
  const float* ff2_w2   = (const float*)d_in[30];
  const float* ff2_b2   = (const float*)d_in[31];
  const float* final_ng = (const float*)d_in[32];
  const float* final_nb = (const float*)d_in[33];

  // workspace carve (all sizes multiple of 256B)
  char* w = (char*)d_ws;
  auto take = [&](size_t bytes) { char* p = w; w += bytes; return p; };
  float*  xbuf   = (float*) take((size_t)Mrows * Dc * 4);        // running residual
  __bf16* abuf   = (__bf16*)take((size_t)Mrows * Dc * 2);        // LN output (bf16 GEMM A)
  __bf16* hbf    = (__bf16*)take((size_t)Mrows * 4 * Dc * 2);    // FFN hidden (bf16)
  float*  t32    = (float*) take((size_t)Mrows * 2 * Dc * 4);    // f32 GEMM scratch (pw1)
  __bf16* qkvbf  = (__bf16*)take((size_t)Mrows * 3 * Dc * 2);
  __bf16* attnbf = (__bf16*)take((size_t)Mrows * Dc * 2);
  float*  glubuf = (float*) take((size_t)Mrows * Dc * 4);
  __bf16* cbuf   = (__bf16*)take((size_t)Mrows * Dc * 2);
  __bf16* wf1a   = (__bf16*)take((size_t)Dc * 4 * Dc * 2);
  __bf16* wf1b   = (__bf16*)take((size_t)4 * Dc * Dc * 2);
  __bf16* wqkvb  = (__bf16*)take((size_t)Dc * 3 * Dc * 2);
  __bf16* woutb  = (__bf16*)take((size_t)Dc * Dc * 2);
  __bf16* relb   = (__bf16*)take((size_t)NHc * 128 * HDc * 2);
  __bf16* wpw1   = (__bf16*)take((size_t)Dc * 2 * Dc * 2);
  __bf16* wpw2   = (__bf16*)take((size_t)Dc * Dc * 2);
  __bf16* wf2a   = (__bf16*)take((size_t)Dc * 4 * Dc * 2);
  __bf16* wf2b   = (__bf16*)take((size_t)4 * Dc * Dc * 2);
  float*  outp   = (float*)d_out;

  auto cvt = [&](const float* src, __bf16* dst, int n) {
    cvt_bf16_kernel<<<(n + 255) / 256, 256, 0, stream>>>(src, dst, n);
  };

  // weight conversion (per call: deterministic, bandwidth-trivial vs. GEMMs)
  cvt(ff1_w1, wf1a, Dc * 4 * Dc);
  cvt(ff1_w2, wf1b, 4 * Dc * Dc);
  cvt(wqkv,   wqkvb, Dc * 3 * Dc);
  cvt(wout,   woutb, Dc * Dc);
  cvt(pw1_w,  wpw1, Dc * 2 * Dc);
  cvt(pw2_w,  wpw2, Dc * Dc);
  cvt(ff2_w1, wf2a, Dc * 4 * Dc);
  cvt(ff2_w2, wf2b, 4 * Dc * Dc);
  relpos_cvt_kernel<<<(128 * 8 * 64 + 255) / 256, 256, 0, stream>>>(rel_pos, relb);

  const dim3 blk256(256);
  const int EW = Mrows * Dc;                    // elementwise count
  const dim3 ewg((EW + 255) / 256);

  // ---- FFN1: x += 0.5 * (silu(ln(x)@w1+b1)@w2+b2), mask-fill ----
  ln_kernel<0><<<Mrows, 256, 0, stream>>>(x, ff1_ng, ff1_nb, abuf, nullptr, nullptr);
  gemm_bf16_kernel<1><<<dim3(4 * Dc / 128, Mrows / 128), blk256, 0, stream>>>(
      abuf, wf1a, ff1_b1, nullptr, nullptr, nullptr, hbf, Mrows, 4 * Dc, Dc);
  gemm_bf16_kernel<3><<<dim3(Dc / 128, Mrows / 128), blk256, 0, stream>>>(
      hbf, wf1b, ff1_b2, x, mask, xbuf, nullptr, Mrows, Dc, 4 * Dc);

  // ---- MHA ----
  ln_kernel<0><<<Mrows, 256, 0, stream>>>(xbuf, mha_ng, mha_nb, abuf, nullptr, nullptr);
  gemm_bf16_kernel<2><<<dim3(3 * Dc / 128, Mrows / 128), blk256, 0, stream>>>(
      abuf, wqkvb, nullptr, nullptr, nullptr, nullptr, qkvbf, Mrows, 3 * Dc, Dc);
  attn_kernel<<<dim3(Tc / 16, NHc, Bc), 32, 0, stream>>>(qkvbf, relb, mask, attnbf);
  gemm_bf16_kernel<4><<<dim3(Dc / 128, Mrows / 128), blk256, 0, stream>>>(
      attnbf, woutb, bout, xbuf, mask, xbuf, nullptr, Mrows, Dc, Dc);

  // ---- Conv module ----
  ln_kernel<0><<<Mrows, 256, 0, stream>>>(xbuf, conv_ng, conv_nb, abuf, nullptr, nullptr);
  gemm_bf16_kernel<0><<<dim3(2 * Dc / 128, Mrows / 128), blk256, 0, stream>>>(
      abuf, wpw1, pw1_b, nullptr, nullptr, t32, nullptr, Mrows, 2 * Dc, Dc);
  glu_kernel<<<ewg, 256, 0, stream>>>(t32, glubuf);
  dwconv_bn_silu_kernel<<<ewg, 256, 0, stream>>>(glubuf, dw_w, dw_b, bn_g, bn_b, bn_m, bn_v, cbuf);
  gemm_bf16_kernel<4><<<dim3(Dc / 128, Mrows / 128), blk256, 0, stream>>>(
      cbuf, wpw2, pw2_b, xbuf, mask, xbuf, nullptr, Mrows, Dc, Dc);

  // ---- FFN2 ----
  ln_kernel<0><<<Mrows, 256, 0, stream>>>(xbuf, ff2_ng, ff2_nb, abuf, nullptr, nullptr);
  gemm_bf16_kernel<1><<<dim3(4 * Dc / 128, Mrows / 128), blk256, 0, stream>>>(
      abuf, wf2a, ff2_b1, nullptr, nullptr, nullptr, hbf, Mrows, 4 * Dc, Dc);
  gemm_bf16_kernel<3><<<dim3(Dc / 128, Mrows / 128), blk256, 0, stream>>>(
      hbf, wf2b, ff2_b2, xbuf, mask, xbuf, nullptr, Mrows, Dc, 4 * Dc);

  // ---- final LN + mask-fill -> d_out (f32) ----
  ln_kernel<1><<<Mrows, 256, 0, stream>>>(xbuf, final_ng, final_nb, nullptr, outp, mask);
}
